// SnakeBlock_83494164234719
// MI455X (gfx1250) — matrix operand order repaired and checked by reading, not compiled, not run on hardware
//
#include <hip/hip_runtime.h>
#include <hip/hip_bf16.h>

typedef __attribute__((ext_vector_type(16))) _Float16 v16h;
typedef __attribute__((ext_vector_type(8)))  float    v8f;
typedef __attribute__((ext_vector_type(4)))  unsigned v4u;
typedef __attribute__((ext_vector_type(8)))  int      v8i;
typedef __attribute__((ext_vector_type(4)))  int      v4i;

#define NIi 512
#define NPi 128
#define BN_INV 0.9999950000374998f

union U8h { uint4 u; _Float16 h[8]; };

__device__ __forceinline__ v16h ldfrag(const _Float16* p0, const _Float16* p1) {
    v16h r;
    ((uint4*)&r)[0] = *(const uint4*)p0;
    ((uint4*)&r)[1] = *(const uint4*)p1;
    return r;
}

// Dual-N K loop: one A (weight) fragment feeds two B (activation) tiles -> 2 wmma per A load.
template<int Ci, int LDX, long long SST>
__device__ __forceinline__ void kloop2(const _Float16* __restrict__ wr,
                                       const _Float16* __restrict__ x0row,
                                       const _Float16* __restrict__ x1row,
                                       int g, v8f& acc0, v8f& acc1) {
#pragma unroll
    for (int kc = 0; kc < Ci; kc += 32) {
        const int stage = kc / LDX;
        const int cc = kc % LDX;
        v16h a = ldfrag(wr + kc + g * 16, wr + kc + g * 16 + 8);
        const _Float16* p0 = x0row + (long long)stage * SST + cc + g * 16;
        const _Float16* p1 = x1row + (long long)stage * SST + cc + g * 16;
        v16h b0 = ldfrag(p0, p0 + 8);
        v16h b1 = ldfrag(p1, p1 + 8);
        acc0 = __builtin_amdgcn_wmma_f32_16x16x32_f16(false, a, false, b0, (short)0, acc0, false, false);
        acc1 = __builtin_amdgcn_wmma_f32_16x16x32_f16(false, a, false, b1, (short)0, acc1, false, false);
    }
}

// ---------------- weight conversion ----------------
__global__ __launch_bounds__(256) void k_cvt_pad(const float* __restrict__ src, _Float16* __restrict__ dst,
                                                 int rows, int scols, int dcols, int soff, int sld) {
    size_t idx = (size_t)blockIdx.x * 256 + threadIdx.x;
    size_t total = (size_t)rows * dcols;
    if (idx >= total) return;
    int c = (int)(idx % dcols);
    int r = (int)(idx / dcols);
    dst[idx] = (c < scols) ? (_Float16)src[(size_t)r * sld + soff + c] : (_Float16)0.f;
}

// src [NB][Co][Ci][9] -> dst [NB][9][Co][Ci]
__global__ __launch_bounds__(256) void k_cvt_tap(const float* __restrict__ src, _Float16* __restrict__ dst,
                                                 int NB, int Co, int Ci) {
    size_t idx = (size_t)blockIdx.x * 256 + threadIdx.x;
    size_t total = (size_t)NB * 9 * Co * Ci;
    if (idx >= total) return;
    int i = (int)(idx % Ci); size_t r = idx / Ci;
    int o = (int)(r % Co); r /= Co;
    int t = (int)(r % 9);  int b = (int)(r / 9);
    dst[idx] = (_Float16)src[(((size_t)b * Co + o) * Ci + i) * 9 + t];
}

// ---------------- grid sample + input assembly ----------------
__global__ __launch_bounds__(128) void k_gather(const float* __restrict__ cnn, const float* __restrict__ poly,
                                                const int* __restrict__ ind, _Float16* __restrict__ X0) {
    __shared__ float sminx[128], sminy[128], smaxx[128], smaxy[128];
    __shared__ float cf[64];
    int inst = blockIdx.x;
    int p = threadIdx.x;
    size_t pb = ((size_t)inst * NPi + p) * 2;
    float px = poly[pb], py = poly[pb + 1];
    sminx[p] = px; smaxx[p] = px; sminy[p] = py; smaxy[p] = py;
    __syncthreads();
    for (int s = 64; s > 0; s >>= 1) {
        if (p < s) {
            sminx[p] = fminf(sminx[p], sminx[p + s]);
            sminy[p] = fminf(sminy[p], sminy[p + s]);
            smaxx[p] = fmaxf(smaxx[p], smaxx[p + s]);
            smaxy[p] = fmaxf(smaxy[p], smaxy[p + s]);
        }
        __syncthreads();
    }
    float minx = sminx[0], miny = sminy[0], maxx = smaxx[0], maxy = smaxy[0];
    const float* fb = cnn + (size_t)ind[inst] * 64 * 16384;
    _Float16* row = X0 + ((size_t)inst * NPi + p) * 160;

    {
        float ix = px - 0.5f, iy = py - 0.5f;
        float x0f = floorf(ix), y0f = floorf(iy);
        float wx = ix - x0f, wy = iy - y0f;
        int x0 = (int)x0f, y0 = (int)y0f, x1 = x0 + 1, y1 = y0 + 1;
        float vx0 = (x0 >= 0 && x0 < 128) ? 1.f : 0.f, vx1 = (x1 >= 0 && x1 < 128) ? 1.f : 0.f;
        float vy0 = (y0 >= 0 && y0 < 128) ? 1.f : 0.f, vy1 = (y1 >= 0 && y1 < 128) ? 1.f : 0.f;
        int cx0 = min(max(x0, 0), 127), cx1 = min(max(x1, 0), 127);
        int cy0 = min(max(y0, 0), 127), cy1 = min(max(y1, 0), 127);
        int o00 = cy0 * 128 + cx0, o01 = cy0 * 128 + cx1, o10 = cy1 * 128 + cx0, o11 = cy1 * 128 + cx1;
        float w00 = (1.f - wx) * (1.f - wy) * vx0 * vy0, w01 = wx * (1.f - wy) * vx1 * vy0;
        float w10 = (1.f - wx) * wy * vx0 * vy1,          w11 = wx * wy * vx1 * vy1;
        for (int c = 0; c < 64; ++c) {
            const float* fc = fb + (size_t)c * 16384;
            float v = w00 * fc[o00] + w01 * fc[o01] + w10 * fc[o10] + w11 * fc[o11];
            row[c] = (_Float16)v;
        }
    }
    row[64] = (_Float16)(px - minx);
    row[65] = (_Float16)(py - miny);

    if (p < 64) {
        float cx = 0.5f * (minx + maxx), cy = 0.5f * (miny + maxy);
        float ix = cx - 0.5f, iy = cy - 0.5f;
        float x0f = floorf(ix), y0f = floorf(iy);
        float wx = ix - x0f, wy = iy - y0f;
        int x0 = (int)x0f, y0 = (int)y0f, x1 = x0 + 1, y1 = y0 + 1;
        float vx0 = (x0 >= 0 && x0 < 128) ? 1.f : 0.f, vx1 = (x1 >= 0 && x1 < 128) ? 1.f : 0.f;
        float vy0 = (y0 >= 0 && y0 < 128) ? 1.f : 0.f, vy1 = (y1 >= 0 && y1 < 128) ? 1.f : 0.f;
        int cx0 = min(max(x0, 0), 127), cx1 = min(max(x1, 0), 127);
        int cy0 = min(max(y0, 0), 127), cy1 = min(max(y1, 0), 127);
        const float* fc = fb + (size_t)p * 16384;
        float v = (1.f - wx) * (1.f - wy) * vx0 * vy0 * fc[cy0 * 128 + cx0]
                + wx * (1.f - wy) * vx1 * vy0 * fc[cy0 * 128 + cx1]
                + (1.f - wx) * wy * vx0 * vy1 * fc[cy1 * 128 + cx0]
                + wx * wy * vx1 * vy1 * fc[cy1 * 128 + cx1];
        cf[p] = v;
    }
    __syncthreads();
    for (int c = 0; c < 64; ++c) row[66 + c] = (_Float16)cf[c];
    for (int c = 130; c < 160; ++c) row[c] = (_Float16)0.f;
}

// ---------------- generic WMMA GEMM ----------------
#define EPI_BIAS 1
#define EPI_RELU 2
#define EPI_GVEC 16

template<int EPI, int Co>
__device__ __forceinline__ void epi_store(const v8f& acc, _Float16* __restrict__ Y, int inst, int l,
                                          int mbase, const float* __restrict__ bias,
                                          const float* __restrict__ gvec) {
    U8h o;
#pragma unroll
    for (int r = 0; r < 8; ++r) {
        float v = acc[r];
        if (EPI & EPI_BIAS) v += bias[mbase + r];
        if (EPI & EPI_GVEC) v += gvec[(size_t)inst * 256 + mbase + r];
        if (EPI & EPI_RELU) v = fmaxf(v, 0.f);
        o.h[r] = (_Float16)v;
    }
    *(uint4*)(Y + ((size_t)inst * NPi + l) * Co + mbase) = o.u;
}

template<int EPI, int Ci, int LDX, long long SST, int Co>
__global__ __launch_bounds__(256) void k_gemm(const _Float16* __restrict__ X, const _Float16* __restrict__ W,
                                              _Float16* __restrict__ Y,
                                              const float* __restrict__ bias, const float* __restrict__ gvec) {
    constexpr int CT = Co >> 4;
    constexpr int CTH = CT / 2;
    int inst = blockIdx.x;
    int tid = threadIdx.x, wv = tid >> 5, lane = tid & 31;
    int g = lane >> 4, nm = lane & 15;
    int ltp = wv & 3, half = wv >> 2;
    int l0 = ltp * 32 + nm;
    const _Float16* x0row = X + ((size_t)inst * NPi + l0) * LDX;
    const _Float16* x1row = x0row + (size_t)16 * LDX;
#pragma unroll
    for (int c = 0; c < CTH; ++c) {
        int ct = half * CTH + c;
        const _Float16* wrp = W + (size_t)(ct * 16 + nm) * Ci;
        v8f acc0 = {0.f, 0.f, 0.f, 0.f, 0.f, 0.f, 0.f, 0.f};
        v8f acc1 = {0.f, 0.f, 0.f, 0.f, 0.f, 0.f, 0.f, 0.f};
        kloop2<Ci, LDX, SST>(wrp, x0row, x1row, g, acc0, acc1);
        int mbase = ct * 16 + 8 * g;
        epi_store<EPI, Co>(acc0, Y, inst, l0, mbase, bias, gvec);
        epi_store<EPI, Co>(acc1, Y, inst, l0 + 16, mbase, bias, gvec);
    }
}

// ---- TDM staging: contiguous global -> LDS copy of (NPi*Ci) f16 via TENSOR_LOAD_TO_LDS ----
__device__ __forceinline__ void tdm_stage(const _Float16* Xi, int ndwords) {
    unsigned long long ga = (unsigned long long)(uintptr_t)Xi;
    v4u g0;
    g0[0] = 1u;                                            // count=1
    g0[1] = 0u;                                            // lds_addr = dynamic LDS base
    g0[2] = (unsigned)ga;
    g0[3] = ((unsigned)(ga >> 32) & 0x01FFFFFFu) | 0x80000000u;  // global_addr hi | type=2
    v8i g1 = {0, 0, 0, 0, 0, 0, 0, 0};
    unsigned nd = (unsigned)ndwords;
    g1[0] = (int)(2u << 16);                               // workgroup_mask=0, data_size=2 (4B)
    g1[1] = (int)((nd & 0xFFFFu) << 16);                   // tensor_dim0[15:0]
    g1[2] = (int)(((nd >> 16) & 0xFFFFu) | (1u << 16));    // tensor_dim0[31:16], tensor_dim1=1
    g1[3] = (int)((nd & 0xFFFFu) << 16);                   // tile_dim0
    g1[5] = (int)nd;                                       // tensor_dim0_stride low
    v4i z4 = {0, 0, 0, 0};
#if defined(__clang_major__) && (__clang_major__ >= 23)
    v8i z8 = {0, 0, 0, 0, 0, 0, 0, 0};
    __builtin_amdgcn_tensor_load_to_lds(g0, g1, z4, z4, z8, 0);
#else
    __builtin_amdgcn_tensor_load_to_lds(g0, g1, z4, z4, 0);
#endif
    __builtin_amdgcn_s_wait_tensorcnt(0);
}

// ---------------- circular dilated conv, k=9, WMMA, TDM-staged LDS input ----------------
template<bool RES, int Ci, int Co>
__global__ __launch_bounds__(256) void k_cconv(const _Float16* __restrict__ X, const _Float16* __restrict__ W,
                                               _Float16* __restrict__ Y,
                                               const float* __restrict__ bias, const float* __restrict__ gamma,
                                               const float* __restrict__ beta, int dil) {
    extern __shared__ _Float16 lds[];
    constexpr int CT = Co >> 4;
    constexpr int CTH = CT / 2;
    int inst = blockIdx.x;
    int tid = threadIdx.x, wv = tid >> 5, lane = tid & 31;
    int g = lane >> 4, nm = lane & 15;
    const _Float16* Xi = X + (size_t)inst * NPi * Ci;
    if (wv == 0) tdm_stage(Xi, (NPi * Ci) >> 1);
    __syncthreads();
    int ltp = wv & 3, half = wv >> 2;
    int l0 = ltp * 32 + nm;
#pragma unroll
    for (int c = 0; c < CTH; ++c) {
        int ct = half * CTH + c;
        v8f acc0 = {0.f, 0.f, 0.f, 0.f, 0.f, 0.f, 0.f, 0.f};
        v8f acc1 = {0.f, 0.f, 0.f, 0.f, 0.f, 0.f, 0.f, 0.f};
#pragma unroll
        for (int t = 0; t < 9; ++t) {
            int sh = (t - 4) * dil + 128;
            int lsrc0 = (l0 + sh) & 127;
            int lsrc1 = (l0 + 16 + sh) & 127;
            const _Float16* wrp = W + (size_t)(t * Co + ct * 16 + nm) * Ci;
            const _Float16* xr0 = lds + (size_t)lsrc0 * Ci;
            const _Float16* xr1 = lds + (size_t)lsrc1 * Ci;
#pragma unroll
            for (int kc = 0; kc < Ci; kc += 32) {
                v16h a = ldfrag(wrp + kc + g * 16, wrp + kc + g * 16 + 8);
                v16h b0 = ldfrag(xr0 + kc + g * 16, xr0 + kc + g * 16 + 8);
                v16h b1 = ldfrag(xr1 + kc + g * 16, xr1 + kc + g * 16 + 8);
                acc0 = __builtin_amdgcn_wmma_f32_16x16x32_f16(false, a, false, b0, (short)0, acc0, false, false);
                acc1 = __builtin_amdgcn_wmma_f32_16x16x32_f16(false, a, false, b1, (short)0, acc1, false, false);
            }
        }
        int mbase = ct * 16 + 8 * g;
#pragma unroll
        for (int pass = 0; pass < 2; ++pass) {
            int l = l0 + pass * 16;
            const v8f& acc = pass ? acc1 : acc0;
            U8h o;
            U8h rsd;
            if (RES) rsd.u = *(const uint4*)(lds + (size_t)l * Ci + mbase);
#pragma unroll
            for (int r = 0; r < 8; ++r) {
                float v = acc[r] + bias[mbase + r];
                v = fmaxf(v, 0.f);
                v = gamma[mbase + r] * BN_INV * v + beta[mbase + r];
                if (RES) v += (float)rsd.h[r];
                o.h[r] = (_Float16)v;
            }
            *(uint4*)(Y + ((size_t)inst * NPi + l) * Co + mbase) = o.u;
        }
    }
}

// ---------------- fusion GEMM (Co=256) + column max over L -> glob[inst][256] f32 ----------------
template<int Ci, int LDX, long long SST>
__global__ __launch_bounds__(256) void k_gemm_colmax(const _Float16* __restrict__ X, const _Float16* __restrict__ W,
                                                     const float* __restrict__ bias, float* __restrict__ glob) {
    __shared__ float pmax[8][256];
    int inst = blockIdx.x;
    int tid = threadIdx.x, wv = tid >> 5, lane = tid & 31;
    int g = lane >> 4, nm = lane & 15;
    for (int i = tid; i < 8 * 256; i += 256) ((float*)pmax)[i] = -3.4e38f;
    __syncthreads();
    int ltp = wv & 3, half = wv >> 2;
    int l0 = ltp * 32 + nm;
    const _Float16* x0row = X + ((size_t)inst * NPi + l0) * LDX;
    const _Float16* x1row = x0row + (size_t)16 * LDX;
    for (int c = 0; c < 8; ++c) {
        int ct = half * 8 + c;
        const _Float16* wrp = W + (size_t)(ct * 16 + nm) * Ci;
        v8f acc0 = {0.f, 0.f, 0.f, 0.f, 0.f, 0.f, 0.f, 0.f};
        v8f acc1 = {0.f, 0.f, 0.f, 0.f, 0.f, 0.f, 0.f, 0.f};
        kloop2<Ci, LDX, SST>(wrp, x0row, x1row, g, acc0, acc1);
#pragma unroll
        for (int r = 0; r < 8; ++r) {
            int mch = ct * 16 + 8 * g + r;
            float v = fmaxf(acc0[r], acc1[r]) + bias[mch];
            for (int off = 8; off; off >>= 1) v = fmaxf(v, __shfl_xor(v, off, 16));
            if (nm == 0) pmax[wv][mch] = v;
        }
    }
    __syncthreads();
    if (tid < 256) {
        float v = pmax[0][tid];
        for (int w = 1; w < 8; ++w) v = fmaxf(v, pmax[w][tid]);
        glob[(size_t)inst * 256 + tid] = v;
    }
}

// ---------------- gvec[inst][m] = pred_w1[m, 0:256] @ glob[inst] + pred_b1[m] ----------------
__global__ __launch_bounds__(256) void k_gvec(const float* __restrict__ glob, const float* __restrict__ w1,
                                              const float* __restrict__ b1, float* __restrict__ gvec) {
    int inst = blockIdx.x, m = threadIdx.x;
    const float* g = glob + (size_t)inst * 256;
    const float* wr = w1 + (size_t)m * 1280;
    float s = b1[m];
    for (int k = 0; k < 256; ++k) s += wr[k] * g[k];
    gvec[(size_t)inst * 256 + m] = s;
}

// ---------------- final 64->2 + output transform ----------------
__global__ __launch_bounds__(256) void k_out(const _Float16* __restrict__ H2, const float* __restrict__ poly,
                                             const float* __restrict__ w3, const float* __restrict__ b3,
                                             float* __restrict__ out) {
    size_t idx = (size_t)blockIdx.x * 256 + threadIdx.x;
    if (idx >= (size_t)NIi * NPi) return;
    const _Float16* h = H2 + idx * 64;
    float a0 = b3[0], a1 = b3[1];
    for (int k = 0; k < 64; ++k) {
        float v = (float)h[k];
        a0 += w3[k] * v;
        a1 += w3[64 + k] * v;
    }
    out[idx * 2 + 0] = poly[idx * 2 + 0] * 4.0f + a0;
    out[idx * 2 + 1] = poly[idx * 2 + 1] * 4.0f + a1;
}

extern "C" void kernel_launch(void* const* d_in, const int* in_sizes, int n_in,
                              void* d_out, int out_size, void* d_ws, size_t ws_size,
                              hipStream_t stream) {
    const float* cnn      = (const float*)d_in[0];
    const float* poly     = (const float*)d_in[1];
    const int*   ind      = (const int*)d_in[2];
    const float* fuse_w1  = (const float*)d_in[3];
    const float* fuse_b1  = (const float*)d_in[4];
    const float* fuse_w2  = (const float*)d_in[5];
    const float* fuse_b2  = (const float*)d_in[6];
    const float* head_w   = (const float*)d_in[7];
    const float* head_b   = (const float*)d_in[8];
    const float* head_g   = (const float*)d_in[9];
    const float* head_bt  = (const float*)d_in[10];
    const float* res_w    = (const float*)d_in[11];
    const float* res_b    = (const float*)d_in[12];
    const float* res_g    = (const float*)d_in[13];
    const float* res_bt   = (const float*)d_in[14];
    const float* fusion_w = (const float*)d_in[15];
    const float* fusion_b = (const float*)d_in[16];
    const float* pred_w1  = (const float*)d_in[17];
    const float* pred_b1  = (const float*)d_in[18];
    const float* pred_w2  = (const float*)d_in[19];
    const float* pred_b2  = (const float*)d_in[20];
    const float* pred_w3  = (const float*)d_in[21];
    const float* pred_b3  = (const float*)d_in[22];
    float* out = (float*)d_out;

    char* p = (char*)d_ws;
    auto alloc = [&](size_t bytes) -> char* {
        char* r = p;
        p += (bytes + 255) & ~(size_t)255;
        return r;
    };
    float*     glob = (float*)alloc((size_t)NIi * 256 * 4);
    float*     gvec = (float*)alloc((size_t)NIi * 256 * 4);
    _Float16*  wf1  = (_Float16*)alloc((size_t)128 * 160 * 2);
    _Float16*  wf2  = (_Float16*)alloc((size_t)64 * 128 * 2);
    _Float16*  wh   = (_Float16*)alloc((size_t)9 * 128 * 64 * 2);
    _Float16*  wr   = (_Float16*)alloc((size_t)7 * 9 * 128 * 128 * 2);
    _Float16*  wfu  = (_Float16*)alloc((size_t)256 * 1024 * 2);
    _Float16*  wp1  = (_Float16*)alloc((size_t)256 * 1024 * 2);
    _Float16*  wp2  = (_Float16*)alloc((size_t)64 * 256 * 2);
    _Float16*  x0   = (_Float16*)alloc((size_t)NIi * NPi * 160 * 2);
    _Float16*  a1   = (_Float16*)alloc((size_t)NIi * NPi * 128 * 2);
    _Float16*  a2   = (_Float16*)alloc((size_t)NIi * NPi * 64 * 2);
    _Float16*  st   = (_Float16*)alloc((size_t)8 * NIi * NPi * 128 * 2);
    _Float16*  h1   = (_Float16*)alloc((size_t)NIi * NPi * 256 * 2);
    _Float16*  h2   = (_Float16*)alloc((size_t)NIi * NPi * 64 * 2);

    constexpr long long SDS = (long long)NIi * NPi * 128;  // state stage stride (elements)
    auto grid1 = [](size_t total) { return dim3((unsigned)((total + 255) / 256)); };

    // --- weight conversion ---
    k_cvt_pad<<<grid1((size_t)128 * 160), 256, 0, stream>>>(fuse_w1, wf1, 128, 130, 160, 0, 130);
    k_cvt_pad<<<grid1((size_t)64 * 128), 256, 0, stream>>>(fuse_w2, wf2, 64, 128, 128, 0, 128);
    k_cvt_tap<<<grid1((size_t)9 * 128 * 64), 256, 0, stream>>>(head_w, wh, 1, 128, 64);
    k_cvt_tap<<<grid1((size_t)7 * 9 * 128 * 128), 256, 0, stream>>>(res_w, wr, 7, 128, 128);
    k_cvt_pad<<<grid1((size_t)256 * 1024), 256, 0, stream>>>(fusion_w, wfu, 256, 1024, 1024, 0, 1024);
    k_cvt_pad<<<grid1((size_t)256 * 1024), 256, 0, stream>>>(pred_w1, wp1, 256, 1024, 1024, 256, 1280);
    k_cvt_pad<<<grid1((size_t)64 * 256), 256, 0, stream>>>(pred_w2, wp2, 64, 256, 256, 0, 256);

    // --- grid sample / input assembly ---
    k_gather<<<NIi, 128, 0, stream>>>(cnn, poly, ind, x0);

    // --- fuse1: 160(pad from 130) -> 128, relu ---
    k_gemm<EPI_BIAS | EPI_RELU, 160, 160, 0, 128><<<NIi, 256, 0, stream>>>(x0, wf1, a1, fuse_b1, nullptr);
    // --- fuse2: 128 -> 64, relu ---
    k_gemm<EPI_BIAS | EPI_RELU, 128, 128, 0, 64><<<NIi, 256, 0, stream>>>(a1, wf2, a2, fuse_b2, nullptr);

    // --- head conv: 64 -> 128, dil 1, relu+bn ---
    k_cconv<false, 64, 128><<<NIi, 256, (size_t)NPi * 64 * 2, stream>>>(a2, wh, st, head_b, head_g, head_bt, 1);

    // --- 7 residual conv blocks ---
    const int DIL[7] = {1, 1, 1, 2, 2, 4, 4};
    for (int i = 0; i < 7; ++i) {
        k_cconv<true, 128, 128><<<NIi, 256, (size_t)NPi * 128 * 2, stream>>>(
            st + (size_t)i * SDS, wr + (size_t)i * 9 * 128 * 128, st + (size_t)(i + 1) * SDS,
            res_b + i * 128, res_g + i * 128, res_bt + i * 128, DIL[i]);
    }

    // --- fusion 1024 -> 256 + max over L ---
    k_gemm_colmax<1024, 128, SDS><<<NIi, 256, 0, stream>>>(st, wfu, fusion_b, glob);
    // --- gvec = pred_w1[:, :256] @ glob + b1 ---
    k_gvec<<<NIi, 256, 0, stream>>>(glob, pred_w1, pred_b1, gvec);
    // --- pred1: states(1024) -> 256, + gvec, relu ---
    k_gemm<EPI_GVEC | EPI_RELU, 1024, 128, SDS, 256><<<NIi, 256, 0, stream>>>(st, wp1, h1, nullptr, gvec);
    // --- pred2: 256 -> 64, relu ---
    k_gemm<EPI_BIAS | EPI_RELU, 256, 256, 0, 64><<<NIi, 256, 0, stream>>>(h1, wp2, h2, pred_b2, nullptr);
    // --- pred3 + output transform ---
    k_out<<<grid1((size_t)NIi * NPi), 256, 0, stream>>>(h2, poly, pred_w3, pred_b3, out);
}
